// AttentionBlock_32238024524154
// MI455X (gfx1250) — matrix operand order repaired and checked by reading, not compiled, not run on hardware
//
#include <hip/hip_runtime.h>
#include <hip/hip_bf16.h>

typedef __bf16 bf16_t;
typedef __attribute__((ext_vector_type(16))) __bf16 v16bf;
typedef __attribute__((ext_vector_type(8)))  float  v8f;
typedef __attribute__((ext_vector_type(4)))  int    v4i_t;

#define IN_DIMC 768
#define DMODEL  384
#define HIDDENC 3072
#define NHEADS  6
#define HEADD   64
#define NQ      1024
#define NCTX    2048
#define NBATCH  8

// ---------------------------------------------------------------------------
// CDNA5 async global->LDS copy (ASYNCcnt) with synchronous fallback.
// Probe-derived signature: (global int4*, lds int4*, int offset, int cpol).
// ---------------------------------------------------------------------------
#if __has_builtin(__builtin_amdgcn_global_load_async_to_lds_b128) && \
    __has_builtin(__builtin_amdgcn_s_wait_asynccnt)
#define USE_ASYNC_LDS 1
#else
#define USE_ASYNC_LDS 0
#endif

// copy 8 bf16 (16 bytes) from global to LDS, per lane
static __device__ __forceinline__ void copy16(const bf16_t* g, bf16_t* l) {
#if USE_ASYNC_LDS
  __builtin_amdgcn_global_load_async_to_lds_b128(
      (__attribute__((address_space(1))) v4i_t*)g,
      (__attribute__((address_space(3))) v4i_t*)l, 0, 0);
#else
  *(uint4*)l = *(const uint4*)g;
#endif
}

static __device__ __forceinline__ void wait_copies() {
#if USE_ASYNC_LDS
  __builtin_amdgcn_s_wait_asynccnt(0);
#endif
}

// ---------------------------------------------------------------------------
// WMMA helpers (CDNA5 wave32, V_WMMA_F32_16X16X32_BF16)
// ---------------------------------------------------------------------------
static __device__ __forceinline__ v8f wmma_bf16(v16bf a, v16bf b, v8f c) {
  return __builtin_amdgcn_wmma_f32_16x16x32_bf16(false, a, false, b, (short)0, c,
                                                 false, false);
}

// A-matrix 16x32 bf16 fragment from row-major source with stride ld.
// lanes 0-15: row=lane, K 0..7 & 16..23 ; lanes 16-31: row=lane-16, K 8..15 & 24..31
static __device__ __forceinline__ v16bf load_a_frag(const bf16_t* base, int ld) {
  const int lane = threadIdx.x & 31;
  const int row  = lane & 15;
  const int kb   = (lane >> 4) << 3;
  const bf16_t* p0 = base + row * ld + kb;
  const bf16_t* p1 = p0 + 16;
  v16bf a;
#pragma unroll
  for (int j = 0; j < 8; ++j) { a[j] = p0[j]; a[8 + j] = p1[j]; }
  return a;
}

// B-matrix 32x16 fragment from a K-contiguous ([N][K]) source with stride ldt.
// lanes 0-15: col=lane, K 0..7 & 16..23 ; lanes 16-31: col=lane-16, K 8..15 & 24..31
static __device__ __forceinline__ v16bf load_bt_frag(const bf16_t* baseT, int ldt) {
  const int lane = threadIdx.x & 31;
  const int col  = lane & 15;
  const int kg   = (lane >> 4) << 3;
  const bf16_t* p = baseT + col * ldt + kg;
  v16bf b;
#pragma unroll
  for (int j = 0; j < 8; ++j) { b[j] = p[j]; b[8 + j] = p[16 + j]; }
  return b;
}

// ---------------------------------------------------------------------------
// Weight conversion fp32[K,N] -> bf16 transposed [N,K] (one-time)
// ---------------------------------------------------------------------------
__global__ void f32_to_bf16_T(const float* __restrict__ in, bf16_t* __restrict__ out,
                              int K, int N) {
  int i = blockIdx.x * 256 + threadIdx.x;
  if (i < K * N) {
    int n = i / K, k = i - n * K;
    out[i] = (bf16_t)in[(size_t)k * N + n];
  }
}

// ---------------------------------------------------------------------------
// LayerNorm over last dim = 768. One 256-thread block per row.
// ---------------------------------------------------------------------------
__global__ __launch_bounds__(256) void layernorm_768(
    const float* __restrict__ x, const float* __restrict__ w,
    const float* __restrict__ b, bf16_t* __restrict__ out_bf,
    float* __restrict__ out_f32) {
  const int row = blockIdx.x;
  const int tid = threadIdx.x;
  const float* xr = x + (size_t)row * IN_DIMC;
  float v[3];
  float s = 0.0f;
#pragma unroll
  for (int i = 0; i < 3; ++i) { v[i] = xr[tid + i * 256]; s += v[i]; }
  __shared__ float red1[8], red2[8];
#pragma unroll
  for (int off = 16; off > 0; off >>= 1) s += __shfl_xor(s, off, 32);
  if ((tid & 31) == 0) red1[tid >> 5] = s;
  __syncthreads();
  float mean = 0.0f;
#pragma unroll
  for (int i = 0; i < 8; ++i) mean += red1[i];
  mean *= (1.0f / IN_DIMC);
  float s2 = 0.0f;
#pragma unroll
  for (int i = 0; i < 3; ++i) { float d = v[i] - mean; s2 += d * d; }
#pragma unroll
  for (int off = 16; off > 0; off >>= 1) s2 += __shfl_xor(s2, off, 32);
  if ((tid & 31) == 0) red2[tid >> 5] = s2;
  __syncthreads();
  float var = 0.0f;
#pragma unroll
  for (int i = 0; i < 8; ++i) var += red2[i];
  var *= (1.0f / IN_DIMC);
  const float inv = rsqrtf(var + 1e-5f);
#pragma unroll
  for (int i = 0; i < 3; ++i) {
    const int c = tid + i * 256;
    const float y = (v[i] - mean) * inv * w[c] + b[c];
    out_bf[(size_t)row * IN_DIMC + c] = (bf16_t)y;
    if (out_f32) out_f32[(size_t)row * IN_DIMC + c] = y;
  }
}

// ---------------------------------------------------------------------------
// Tiled bf16 GEMM: C[M,N] = act(A[M,K] @ Bt[N,K]^T + bias) (+resid)
// Block 128x128, 8 waves (4x2), wave tile 32x64, K-step 32,
// double-buffered LDS fed by async global->LDS copies.
// KVSPLIT epilogue scatters K to [b,h,m,d] and V (transposed) to [b,h,d,m].
// ---------------------------------------------------------------------------
template <bool OUT_BF16, bool GELU_ACT, bool RESID, bool KVSPLIT = false>
__global__ __launch_bounds__(256) void gemm_bf16(
    const bf16_t* __restrict__ A, const bf16_t* __restrict__ Bt,
    const float* __restrict__ bias, const float* __restrict__ resid,
    void* __restrict__ Cout, int M, int N, int K,
    bf16_t* __restrict__ kdst, bf16_t* __restrict__ vdst, int m_shift) {
  __shared__ __align__(16) bf16_t sA[2][128 * 32];   // [m][k]
  __shared__ __align__(16) bf16_t sBt[2][128 * 32];  // [n][k]

  const int tid  = threadIdx.x;
  const int wave = tid >> 5;
  const int lane = tid & 31;
  const int m0 = blockIdx.x * 128;
  const int n0 = blockIdx.y * 128;
  const int wm = (wave & 3) * 32;
  const int wn = (wave >> 2) * 64;

  v8f acc[2][4] = {};

  const int cr = tid >> 1, cs = (tid & 1) * 16;  // tile copy: 128 rows x 32 cols

  const bf16_t* gA = A  + (size_t)(m0 + cr) * K + cs;
  const bf16_t* gB = Bt + (size_t)(n0 + cr) * K + cs;

  // prologue: tile 0 -> buffer 0
  copy16(gA,     sA[0] + cr * 32 + cs);
  copy16(gA + 8, sA[0] + cr * 32 + cs + 8);
  copy16(gB,     sBt[0] + cr * 32 + cs);
  copy16(gB + 8, sBt[0] + cr * 32 + cs + 8);
  wait_copies();
  __syncthreads();

  const int nsteps = K >> 5;
  for (int it = 0; it < nsteps; ++it) {
    const int cur = it & 1;
    if (it + 1 < nsteps) {  // prefetch next tile into the other buffer
      const bf16_t* nA = gA + (it + 1) * 32;
      const bf16_t* nB = gB + (it + 1) * 32;
      copy16(nA,     sA[cur ^ 1] + cr * 32 + cs);
      copy16(nA + 8, sA[cur ^ 1] + cr * 32 + cs + 8);
      copy16(nB,     sBt[cur ^ 1] + cr * 32 + cs);
      copy16(nB + 8, sBt[cur ^ 1] + cr * 32 + cs + 8);
    }
    v16bf af0 = load_a_frag(sA[cur] + wm * 32, 32);
    v16bf af1 = load_a_frag(sA[cur] + (wm + 16) * 32, 32);
#pragma unroll
    for (int nt = 0; nt < 4; ++nt) {
      v16bf bf = load_bt_frag(sBt[cur] + (wn + nt * 16) * 32, 32);
      acc[0][nt] = wmma_bf16(af0, bf, acc[0][nt]);
      acc[1][nt] = wmma_bf16(af1, bf, acc[1][nt]);
    }
    wait_copies();
    __syncthreads();
  }

  const int col = lane & 15;
  const int rb  = (lane >> 4) << 3;
#pragma unroll
  for (int mt = 0; mt < 2; ++mt) {
#pragma unroll
    for (int nt = 0; nt < 4; ++nt) {
      const int gn = n0 + wn + nt * 16 + col;
      if (KVSPLIT) {
        const int sel = gn >= DMODEL;
        const int dd  = gn & 63;
        const int hh2 = ((gn - sel * DMODEL) >> 6);
        const size_t seqm = (size_t)1 << m_shift;
#pragma unroll
        for (int r = 0; r < 8; ++r) {
          const int gm = m0 + wm + mt * 16 + rb + r;
          const int b  = gm >> m_shift;
          const int mm = gm & ((1 << m_shift) - 1);
          const bf16_t val = (bf16_t)acc[mt][nt][r];
          if (!sel)
            kdst[(((size_t)b * NHEADS + hh2) * seqm + mm) * HEADD + dd] = val;
          else
            vdst[(((size_t)b * NHEADS + hh2) * HEADD + dd) * seqm + mm] = val;
        }
      } else {
        const float bv = bias ? bias[gn] : 0.0f;
#pragma unroll
        for (int r = 0; r < 8; ++r) {
          const size_t gi = (size_t)(m0 + wm + mt * 16 + rb + r) * N + gn;
          float x = acc[mt][nt][r] + bv;
          if (GELU_ACT) x = 0.5f * x * (1.0f + erff(x * 0.70710678118f));
          if (RESID) x += resid[gi];
          if (OUT_BF16) ((bf16_t*)Cout)[gi] = (bf16_t)x;
          else          ((float*)Cout)[gi]  = x;
        }
      }
    }
  }
}

// ---------------------------------------------------------------------------
// Flash attention: softmax(Q K^T / 8) V, head dim 64, bf16 WMMA.
// K stored [b,h,m,64]; V stored transposed [b,h,64,m] -> all tile copies are
// contiguous b128 async copies, all fragment reads are ds_load_b128.
// Grid (NQ/128, NHEADS, NBATCH); 8 waves, 16 q rows each; double-buffered.
// ---------------------------------------------------------------------------
__global__ __launch_bounds__(256) void flash_attn(
    const bf16_t* __restrict__ Q,    // [B, NQ, DMODEL]
    const bf16_t* __restrict__ Kb,   // [B, H, m, 64]
    const bf16_t* __restrict__ Vtb,  // [B, H, 64, m]
    bf16_t* __restrict__ Oo,         // [B, NQ, DMODEL]
    int seq_m) {
  __shared__ __align__(16) bf16_t sK[2][64 * 64];   // [key][feat]
  __shared__ __align__(16) bf16_t sVt[2][64 * 64];  // [feat][key]
  __shared__ __align__(16) bf16_t sP[8][16 * 64];   // per-wave P tile

  const int tid = threadIdx.x, wave = tid >> 5, lane = tid & 31;
  const int bb = blockIdx.z, hh = blockIdx.y;
  const int n0 = blockIdx.x * 128 + wave * 16;
  const float scale = 0.125f;

  const bf16_t* Kh  = Kb  + ((size_t)bb * NHEADS + hh) * seq_m * HEADD;
  const bf16_t* Vth = Vtb + ((size_t)bb * NHEADS + hh) * HEADD * seq_m;

  const bf16_t* qbase = Q + ((size_t)bb * NQ + n0) * DMODEL + hh * HEADD;
  v16bf qa0 = load_a_frag(qbase, DMODEL);
  v16bf qa1 = load_a_frag(qbase + 32, DMODEL);

  v8f oacc[4] = {};
  float m_run[8], l_run[8];
#pragma unroll
  for (int r = 0; r < 8; ++r) { m_run[r] = -1e30f; l_run[r] = 0.0f; }

  const int trow = tid >> 2, tseg = (tid & 3) * 16;  // 64 rows x 64 cols copies
  const int col = lane & 15, rbase = (lane >> 4) << 3;

  // prologue: chunk 0 -> buffer 0
  copy16(Kh + (size_t)trow * HEADD + tseg,      sK[0] + trow * 64 + tseg);
  copy16(Kh + (size_t)trow * HEADD + tseg + 8,  sK[0] + trow * 64 + tseg + 8);
  copy16(Vth + (size_t)trow * seq_m + tseg,     sVt[0] + trow * 64 + tseg);
  copy16(Vth + (size_t)trow * seq_m + tseg + 8, sVt[0] + trow * 64 + tseg + 8);
  wait_copies();
  __syncthreads();

  const int niter = seq_m >> 6;
  for (int it = 0; it < niter; ++it) {
    const int cur = it & 1;
    if (it + 1 < niter) {  // prefetch next 64-key chunk
      const int mc = (it + 1) << 6;
      copy16(Kh + (size_t)(mc + trow) * HEADD + tseg,     sK[cur ^ 1] + trow * 64 + tseg);
      copy16(Kh + (size_t)(mc + trow) * HEADD + tseg + 8, sK[cur ^ 1] + trow * 64 + tseg + 8);
      copy16(Vth + (size_t)trow * seq_m + mc + tseg,      sVt[cur ^ 1] + trow * 64 + tseg);
      copy16(Vth + (size_t)trow * seq_m + mc + tseg + 8,  sVt[cur ^ 1] + trow * 64 + tseg + 8);
    }

    // S = Q @ K^T
    v8f sfr[4];
#pragma unroll
    for (int nt = 0; nt < 4; ++nt) {
      v8f z = {};
      v16bf b0 = load_bt_frag(sK[cur] + nt * 16 * 64, 64);
      v16bf b1 = load_bt_frag(sK[cur] + nt * 16 * 64 + 32, 64);
      z = wmma_bf16(qa0, b0, z);
      z = wmma_bf16(qa1, b1, z);
      sfr[nt] = z;
    }

    // online softmax; each lane owns rows rbase..rbase+7 of the 16-row tile
#pragma unroll
    for (int r = 0; r < 8; ++r) {
      float mx = -1e30f;
#pragma unroll
      for (int nt = 0; nt < 4; ++nt) {
        float v = sfr[nt][r] * scale;
        sfr[nt][r] = v;
        mx = fmaxf(mx, v);
      }
#pragma unroll
      for (int off = 1; off < 16; off <<= 1) mx = fmaxf(mx, __shfl_xor(mx, off, 32));
      const float mnew  = fmaxf(m_run[r], mx);
      const float alpha = __expf(m_run[r] - mnew);
      m_run[r] = mnew;
      l_run[r] *= alpha;
#pragma unroll
      for (int dt = 0; dt < 4; ++dt) oacc[dt][r] *= alpha;
      float rs = 0.0f;
#pragma unroll
      for (int nt = 0; nt < 4; ++nt) {
        float p = __expf(sfr[nt][r] - mnew);
        sfr[nt][r] = p;
        rs += p;
      }
#pragma unroll
      for (int off = 1; off < 16; off <<= 1) rs += __shfl_xor(rs, off, 32);
      l_run[r] += rs;
    }

    // P (bf16) -> per-wave LDS tile, refragment as A operand
#pragma unroll
    for (int nt = 0; nt < 4; ++nt)
#pragma unroll
      for (int r = 0; r < 8; ++r)
        sP[wave][(rbase + r) * 64 + nt * 16 + col] = (bf16_t)sfr[nt][r];
    __syncthreads();

    v16bf pa0 = load_a_frag(&sP[wave][0], 64);
    v16bf pa1 = load_a_frag(&sP[wave][0] + 32, 64);
#pragma unroll
    for (int dt = 0; dt < 4; ++dt) {
      v16bf v0 = load_bt_frag(sVt[cur] + dt * 16 * 64, 64);
      v16bf v1 = load_bt_frag(sVt[cur] + dt * 16 * 64 + 32, 64);
      oacc[dt] = wmma_bf16(pa0, v0, oacc[dt]);
      oacc[dt] = wmma_bf16(pa1, v1, oacc[dt]);
    }

    wait_copies();     // prefetched chunk landed
    __syncthreads();   // everyone done reading `cur`
  }

  bf16_t* obase = Oo + ((size_t)bb * NQ + n0) * DMODEL + hh * HEADD;
#pragma unroll
  for (int dt = 0; dt < 4; ++dt)
#pragma unroll
    for (int r = 0; r < 8; ++r) {
      const float val = oacc[dt][r] / l_run[r];
      obase[(rbase + r) * DMODEL + dt * 16 + col] = (bf16_t)val;
    }
}

// ---------------------------------------------------------------------------
// Host orchestration
// ---------------------------------------------------------------------------
extern "C" void kernel_launch(void* const* d_in, const int* in_sizes, int n_in,
                              void* d_out, int out_size, void* d_ws, size_t ws_size,
                              hipStream_t stream) {
  const float* query   = (const float*)d_in[0];
  const float* context = (const float*)d_in[1];
  const float* ln_w    = (const float*)d_in[2];
  const float* ln_b    = (const float*)d_in[3];
  const float* a1_wq   = (const float*)d_in[4];
  const float* a1_wkv  = (const float*)d_in[5];
  const float* a1_wp   = (const float*)d_in[6];
  const float* a1_bp   = (const float*)d_in[7];
  const float* m1_w1   = (const float*)d_in[8];
  const float* m1_b1   = (const float*)d_in[9];
  const float* m1_w2   = (const float*)d_in[10];
  const float* m1_b2   = (const float*)d_in[11];
  const float* a2_wq   = (const float*)d_in[12];
  const float* a2_wkv  = (const float*)d_in[13];
  const float* a2_wp   = (const float*)d_in[14];
  const float* a2_bp   = (const float*)d_in[15];
  const float* m2_w1   = (const float*)d_in[16];
  const float* m2_b1   = (const float*)d_in[17];
  const float* m2_w2   = (const float*)d_in[18];
  const float* m2_b2   = (const float*)d_in[19];

  const size_t NTOK  = (size_t)NBATCH * NQ;    // 8192
  const size_t NCTOK = (size_t)NBATCH * NCTX;  // 16384

  float* out_c = (float*)d_out;
  float* out_q = (float*)d_out + NTOK * IN_DIMC;

  char* ws = (char*)d_ws;
  size_t off = 0;
  auto alloc = [&](size_t bytes) -> void* {
    void* p = ws + off;
    off = (off + bytes + 255) & ~(size_t)255;
    return p;
  };

  // transposed bf16 weights [N][K]
  bf16_t* wq1t  = (bf16_t*)alloc((size_t)DMODEL * IN_DIMC * 2);
  bf16_t* wkv1t = (bf16_t*)alloc((size_t)2 * DMODEL * IN_DIMC * 2);
  bf16_t* wp1t  = (bf16_t*)alloc((size_t)IN_DIMC * DMODEL * 2);
  bf16_t* w11t  = (bf16_t*)alloc((size_t)HIDDENC * IN_DIMC * 2);
  bf16_t* w21t  = (bf16_t*)alloc((size_t)IN_DIMC * HIDDENC * 2);
  bf16_t* wq2t  = (bf16_t*)alloc((size_t)DMODEL * IN_DIMC * 2);
  bf16_t* wkv2t = (bf16_t*)alloc((size_t)2 * DMODEL * IN_DIMC * 2);
  bf16_t* wp2t  = (bf16_t*)alloc((size_t)IN_DIMC * DMODEL * 2);
  bf16_t* w12t  = (bf16_t*)alloc((size_t)HIDDENC * IN_DIMC * 2);
  bf16_t* w22t  = (bf16_t*)alloc((size_t)IN_DIMC * HIDDENC * 2);

  bf16_t* qln  = (bf16_t*)alloc(NTOK * IN_DIMC * 2);
  bf16_t* lnc  = (bf16_t*)alloc(NTOK * IN_DIMC * 2);
  bf16_t* Qb   = (bf16_t*)alloc(NTOK * DMODEL * 2);
  bf16_t* Kbuf = (bf16_t*)alloc((size_t)NBATCH * NHEADS * NCTX * HEADD * 2);
  bf16_t* Vtb  = (bf16_t*)alloc((size_t)NBATCH * NHEADS * NCTX * HEADD * 2);
  bf16_t* Ab   = (bf16_t*)alloc(NTOK * DMODEL * 2);
  float*  cbuf = (float*)alloc(NTOK * IN_DIMC * 4);
  bf16_t* hbuf = (bf16_t*)alloc(NTOK * HIDDENC * 2);
  bf16_t* cln  = (bf16_t*)alloc(NCTOK * IN_DIMC * 2);

  auto cvtT = [&](const float* s, bf16_t* d, int K, int N) {
    f32_to_bf16_T<<<(K * N + 255) / 256, 256, 0, stream>>>(s, d, K, N);
  };
  cvtT(a1_wq, wq1t, IN_DIMC, DMODEL);
  cvtT(a1_wkv, wkv1t, IN_DIMC, 2 * DMODEL);
  cvtT(a1_wp, wp1t, DMODEL, IN_DIMC);
  cvtT(m1_w1, w11t, IN_DIMC, HIDDENC);
  cvtT(m1_w2, w21t, HIDDENC, IN_DIMC);
  cvtT(a2_wq, wq2t, IN_DIMC, DMODEL);
  cvtT(a2_wkv, wkv2t, IN_DIMC, 2 * DMODEL);
  cvtT(a2_wp, wp2t, DMODEL, IN_DIMC);
  cvtT(m2_w1, w12t, IN_DIMC, HIDDENC);
  cvtT(m2_w2, w22t, HIDDENC, IN_DIMC);

  const dim3 blk(256);

  layernorm_768<<<(int)NTOK, 256, 0, stream>>>(query, ln_w, ln_b, qln, out_q);
  layernorm_768<<<(int)NCTOK, 256, 0, stream>>>(context, ln_w, ln_b, cln, nullptr);

  // ---- Block 1: cross attention (m = 2048, m_shift = 11) ----
  gemm_bf16<true, false, false><<<dim3(64, 3), blk, 0, stream>>>(
      qln, wq1t, nullptr, nullptr, Qb, (int)NTOK, DMODEL, IN_DIMC, nullptr, nullptr, 0);
  gemm_bf16<false, false, false, true><<<dim3(128, 6), blk, 0, stream>>>(
      cln, wkv1t, nullptr, nullptr, nullptr, (int)NCTOK, 2 * DMODEL, IN_DIMC,
      Kbuf, Vtb, 11);
  flash_attn<<<dim3(NQ / 128, NHEADS, NBATCH), blk, 0, stream>>>(Qb, Kbuf, Vtb, Ab, NCTX);
  gemm_bf16<false, false, false><<<dim3(64, 6), blk, 0, stream>>>(
      Ab, wp1t, a1_bp, nullptr, cbuf, (int)NTOK, IN_DIMC, DMODEL, nullptr, nullptr, 0);

  // ---- Block 1 MLP ----
  layernorm_768<<<(int)NTOK, 256, 0, stream>>>(cbuf, ln_w, ln_b, lnc, nullptr);
  gemm_bf16<true, true, false><<<dim3(64, 24), blk, 0, stream>>>(
      lnc, w11t, m1_b1, nullptr, hbuf, (int)NTOK, HIDDENC, IN_DIMC, nullptr, nullptr, 0);
  gemm_bf16<false, false, true><<<dim3(64, 6), blk, 0, stream>>>(
      hbuf, w21t, m1_b2, cbuf, cbuf, (int)NTOK, IN_DIMC, HIDDENC, nullptr, nullptr, 0);

  // ---- Block 2: cross attention, context = ln(c) (m = 1024, m_shift = 10) ----
  layernorm_768<<<(int)NTOK, 256, 0, stream>>>(cbuf, ln_w, ln_b, lnc, nullptr);
  gemm_bf16<true, false, false><<<dim3(64, 3), blk, 0, stream>>>(
      qln, wq2t, nullptr, nullptr, Qb, (int)NTOK, DMODEL, IN_DIMC, nullptr, nullptr, 0);
  gemm_bf16<false, false, false, true><<<dim3(64, 6), blk, 0, stream>>>(
      lnc, wkv2t, nullptr, nullptr, nullptr, (int)NTOK, 2 * DMODEL, IN_DIMC,
      Kbuf, Vtb, 10);
  flash_attn<<<dim3(NQ / 128, NHEADS, NBATCH), blk, 0, stream>>>(Qb, Kbuf, Vtb, Ab, NQ);
  gemm_bf16<false, false, true><<<dim3(64, 6), blk, 0, stream>>>(
      Ab, wp2t, a2_bp, cbuf, cbuf, (int)NTOK, IN_DIMC, DMODEL, nullptr, nullptr, 0);

  // ---- Block 2 MLP -> out_c ----
  layernorm_768<<<(int)NTOK, 256, 0, stream>>>(cbuf, ln_w, ln_b, lnc, nullptr);
  gemm_bf16<true, true, false><<<dim3(64, 24), blk, 0, stream>>>(
      lnc, w12t, m2_b1, nullptr, hbuf, (int)NTOK, HIDDENC, IN_DIMC, nullptr, nullptr, 0);
  gemm_bf16<false, false, true><<<dim3(64, 6), blk, 0, stream>>>(
      hbuf, w22t, m2_b2, cbuf, out_c, (int)NTOK, IN_DIMC, HIDDENC, nullptr, nullptr, 0);
}
